// CrossAttention_33870112096277
// MI455X (gfx1250) — matrix operand order repaired and checked by reading, not compiled, not run on hardware
//
#include <hip/hip_runtime.h>
#include <hip/hip_bf16.h>

// ---------------------------------------------------------------------------
// CrossAttention for MI455X (gfx1250, wave32, WMMA + async-to-LDS)
//   text:(4,512,1024)  av:(4,16,64,1024)  W*:(1024,1024)  idx:(4,512)
//   out = [ (4,512,1024) attention output ; (4,512,64) attn head-mean ]
// Pipeline:
//   1. Wq/Wk/Wv  f32 -> bf16 repack
//   2. LayerNorm(text)->bf16, LayerNorm(av)->bf16
//   3. projections via v_wmma_f32_16x16x32_bf16; A and B tiles staged in LDS
//      with GLOBAL_LOAD_ASYNC_TO_LDS_B128, QUAD-buffered, one barrier per
//      K-step; async INST_OFFSET offsets both LDS+global sides so the steady
//      state has no address arithmetic (pre-biased LDS dest pointers).
//   4. per-word gathered sigmoid attention (f32 VALU, L2-resident k/v)
// ---------------------------------------------------------------------------

typedef __attribute__((ext_vector_type(16))) __bf16 v16bf;
typedef __attribute__((ext_vector_type(8)))  float  v8f;

union BF16Frag { uint4 u[2]; v16bf v; };

#define GLOBAL_AS __attribute__((address_space(1)))
#define LDS_AS    __attribute__((address_space(3)))

// Builtin's pointee is a GCC-vector int4 (per hipcc diagnostic)
typedef int tdm_int4 __attribute__((vector_size(16)));

#if __has_builtin(__builtin_amdgcn_global_load_async_to_lds_b128)
#define HAVE_ASYNC_LDS 1
#else
#define HAVE_ASYNC_LDS 0
#endif

template <int OFF>
__device__ __forceinline__ void async_copy_b128(const void* g, void* l) {
#if HAVE_ASYNC_LDS
    // HW: OFF is added to both the LDS and the global address
    __builtin_amdgcn_global_load_async_to_lds_b128(
        (GLOBAL_AS tdm_int4*)g, (LDS_AS tdm_int4*)l, OFF, /*cpol=*/0);
#else
    *reinterpret_cast<uint4*>((char*)l + OFF) =
        *reinterpret_cast<const uint4*>((const char*)g + OFF);
#endif
}

template <int N>
__device__ __forceinline__ void wait_async() {
#if __has_builtin(__builtin_amdgcn_s_wait_asynccnt)
    __builtin_amdgcn_s_wait_asynccnt((unsigned short)N);
#elif HAVE_ASYNC_LDS
    asm volatile("s_wait_asynccnt %0" :: "i"(N) : "memory");
#endif
}

__device__ __forceinline__ unsigned short f32_to_bf16_rne(float f) {
    unsigned u = __float_as_uint(f);
    unsigned r = u + 0x7FFFu + ((u >> 16) & 1u);   // round to nearest even
    return (unsigned short)(r >> 16);
}

// ---------------- f32 -> bf16 repack (weights) -----------------------------
__global__ void __launch_bounds__(256)
cvt_f32_bf16_kernel(const float* __restrict__ src, unsigned short* __restrict__ dst, int n4) {
    int i = blockIdx.x * 256 + threadIdx.x;
    if (i >= n4) return;
    float4 v = reinterpret_cast<const float4*>(src)[i];
    unsigned lo = (unsigned)f32_to_bf16_rne(v.x) | ((unsigned)f32_to_bf16_rne(v.y) << 16);
    unsigned hi = (unsigned)f32_to_bf16_rne(v.z) | ((unsigned)f32_to_bf16_rne(v.w) << 16);
    reinterpret_cast<uint2*>(dst)[i] = make_uint2(lo, hi);
}

// ---------------- LayerNorm(row of 1024) -> bf16 ---------------------------
__global__ void __launch_bounds__(256)
layernorm_bf16_kernel(const float* __restrict__ x,
                      const float* __restrict__ gamma,
                      const float* __restrict__ beta,
                      unsigned short* __restrict__ y) {
    const int row = blockIdx.x;
    const int t   = threadIdx.x;              // 0..255, 4 elems each
    const float4* xr = reinterpret_cast<const float4*>(x + (size_t)row * 1024);
    float4 v = xr[t];
    float s  = v.x + v.y + v.z + v.w;
    float ss = v.x * v.x + v.y * v.y + v.z * v.z + v.w * v.w;

    __shared__ float rs1[256];
    __shared__ float rs2[256];
    rs1[t] = s; rs2[t] = ss;
    __syncthreads();
    #pragma unroll
    for (int off = 128; off > 0; off >>= 1) {
        if (t < off) { rs1[t] += rs1[t + off]; rs2[t] += rs2[t + off]; }
        __syncthreads();
    }
    const float mu  = rs1[0] * (1.0f / 1024.0f);
    const float var = rs2[0] * (1.0f / 1024.0f) - mu * mu;
    const float rsig = rsqrtf(var + 1e-5f);

    float4 g = reinterpret_cast<const float4*>(gamma)[t];
    float4 b = reinterpret_cast<const float4*>(beta)[t];
    float o0 = (v.x - mu) * rsig * g.x + b.x;
    float o1 = (v.y - mu) * rsig * g.y + b.y;
    float o2 = (v.z - mu) * rsig * g.z + b.z;
    float o3 = (v.w - mu) * rsig * g.w + b.w;
    unsigned lo = (unsigned)f32_to_bf16_rne(o0) | ((unsigned)f32_to_bf16_rne(o1) << 16);
    unsigned hi = (unsigned)f32_to_bf16_rne(o2) | ((unsigned)f32_to_bf16_rne(o3) << 16);
    reinterpret_cast<uint2*>(y + (size_t)row * 1024)[t] = make_uint2(lo, hi);
}

// ---------------- GEMM: out[M,1024] = A[M,1024] @ W^T + bias ---------------
// Block = 256 threads = 8 waves; computes 128(M) x 64(N).
// Per K-step (32): A tile 128x32 bf16 + B tile 64x32 bf16 staged via async,
// 4 LDS buffers, outer loop of 8 with 4 hand-unrolled steps (period == 4).
#define BROW 40      // shorts per LDS tile row (64B data + 16B pad)

__global__ void __launch_bounds__(256)
gemm_bf16_wmma_kernel(const unsigned short* __restrict__ A,
                      const unsigned short* __restrict__ W,
                      const float* __restrict__ bias,
                      float* __restrict__ out) {
    const int K = 1024, N = 1024;
    const int t    = threadIdx.x;
    const int lane = t & 31;
    const int wave = t >> 5;
    const int mblk   = blockIdx.x * 128;
    const int n_base = blockIdx.y * 64;

    __shared__ unsigned short atile[4][128 * BROW];  // 4 x 10240 B
    __shared__ unsigned short btile[4][64 * BROW];   // 4 x  5120 B

    // staging roles: thread t copies 16B chunks: row = t>>2, chunk c = t&3
    const int srow = t >> 2;        // 0..63
    const int c8   = (t & 3) << 3;  // 0,8,16,24 shorts
    const unsigned short* gA0 = A + (size_t)(mblk + srow) * K + c8;        // rows 0..63
    const unsigned short* gA1 = gA0 + (size_t)64 * K;                      // rows 64..127
    const unsigned short* gB  = W + (size_t)(n_base + srow) * K + c8;
    const int lAoff0 = srow * BROW + c8;
    const int lAoff1 = (srow + 64) * BROW + c8;
    const int lBoff  = srow * BROW + c8;

    // LDS destinations for inner step J target buffer (J+1)&3 with HW adding
    // (J+1)*64B to both sides -> pre-bias pointers by -(J+1)*32 shorts.
    unsigned short* a0dst[4]; unsigned short* a1dst[4]; unsigned short* bdst[4];
    #pragma unroll
    for (int j = 0; j < 4; ++j) {
        const int buf  = (j + 1) & 3;
        const int bias_s = (j + 1) * 32;   // shorts == (j+1)*64 bytes
        a0dst[j] = &atile[buf][lAoff0] - bias_s;
        a1dst[j] = &atile[buf][lAoff1] - bias_s;
        bdst[j]  = &btile[buf][lBoff]  - bias_s;
    }

    // fragment read offsets (shorts)
    const int afoff = (wave * 16 + (lane & 15)) * BROW + ((lane >> 4) << 4);
    const int bfoff = (lane & 15) * BROW + ((lane >> 4) << 4);

    v8f acc0 = {}, acc1 = {}, acc2 = {}, acc3 = {};

    // prologue: stage K-step 0 into buffer 0
    async_copy_b128<0>(gA0, &atile[0][lAoff0]);
    async_copy_b128<0>(gA1, &atile[0][lAoff1]);
    async_copy_b128<0>(gB,  &btile[0][lBoff]);

#define STAGE(J)                                                               \
    async_copy_b128<(J + 1) * 64>(gA0, a0dst[J]);                              \
    async_copy_b128<(J + 1) * 64>(gA1, a1dst[J]);                              \
    async_copy_b128<(J + 1) * 64>(gB,  bdst[J]);

#define COMPUTE(BUF)                                                           \
    {                                                                          \
        const unsigned short* at = atile[BUF];                                 \
        const unsigned short* bt = btile[BUF];                                 \
        BF16Frag a, b0, b1, b2, b3;                                            \
        a.u[0]  = *reinterpret_cast<const uint4*>(at + afoff);                 \
        a.u[1]  = *reinterpret_cast<const uint4*>(at + afoff + 8);             \
        b0.u[0] = *reinterpret_cast<const uint4*>(bt + bfoff);                 \
        b0.u[1] = *reinterpret_cast<const uint4*>(bt + bfoff + 8);             \
        b1.u[0] = *reinterpret_cast<const uint4*>(bt + bfoff + 16 * BROW);     \
        b1.u[1] = *reinterpret_cast<const uint4*>(bt + bfoff + 16 * BROW + 8); \
        b2.u[0] = *reinterpret_cast<const uint4*>(bt + bfoff + 32 * BROW);     \
        b2.u[1] = *reinterpret_cast<const uint4*>(bt + bfoff + 32 * BROW + 8); \
        b3.u[0] = *reinterpret_cast<const uint4*>(bt + bfoff + 48 * BROW);     \
        b3.u[1] = *reinterpret_cast<const uint4*>(bt + bfoff + 48 * BROW + 8); \
        acc0 = __builtin_amdgcn_wmma_f32_16x16x32_bf16(false, a.v, false, b0.v, (short)0, acc0, false, false); \
        acc1 = __builtin_amdgcn_wmma_f32_16x16x32_bf16(false, a.v, false, b1.v, (short)0, acc1, false, false); \
        acc2 = __builtin_amdgcn_wmma_f32_16x16x32_bf16(false, a.v, false, b2.v, (short)0, acc2, false, false); \
        acc3 = __builtin_amdgcn_wmma_f32_16x16x32_bf16(false, a.v, false, b3.v, (short)0, acc3, false, false); \
    }

    #pragma unroll 1
    for (int o = 0; o < 8; ++o) {
        STAGE(0); wait_async<3>(); __syncthreads(); COMPUTE(0);
        STAGE(1); wait_async<3>(); __syncthreads(); COMPUTE(1);
        STAGE(2); wait_async<3>(); __syncthreads(); COMPUTE(2);
        if (o < 7) { STAGE(3); wait_async<3>(); } else { wait_async<0>(); }
        __syncthreads(); COMPUTE(3);
        gA0 += 128; gA1 += 128; gB += 128;   // advance 256B (4 K-steps)
    }
#undef STAGE
#undef COMPUTE

    // C/D layout: lanes 0-15: VGPR j <-> M=j; lanes 16-31: VGPR j <-> M=8+j
    const int nl  = lane & 15;
    const int mr0 = (lane >> 4) << 3;
    const int m0  = mblk + wave * 16;
    v8f accs[4] = {acc0, acc1, acc2, acc3};
    #pragma unroll
    for (int nt = 0; nt < 4; ++nt) {
        const int n = n_base + nt * 16 + nl;
        const float bv = bias[n];
        float* op = out + (size_t)(m0 + mr0) * N + n;
        #pragma unroll
        for (int j = 0; j < 8; ++j) op[(size_t)j * N] = accs[nt][j] + bv;
    }
}

// ---------------- gathered sigmoid attention -------------------------------
// One block per word (bs*num_word = 2048). q:(2048,1024) k,v:(4096,1024) f32.
__global__ void __launch_bounds__(256)
attention_kernel(const float* __restrict__ q,
                 const float* __restrict__ k,
                 const float* __restrict__ v,
                 const int*   __restrict__ sidx,
                 float* __restrict__ out,
                 float* __restrict__ attn_mean) {
    const int word = blockIdx.x;           // b*512 + w
    const int b    = word >> 9;
    const int s    = sidx[word];
    const float* qr = q + (size_t)word * 1024;
    const float* kr = k + ((size_t)(b * 16 + s)) * 64 * 1024;
    const float* vr = v + ((size_t)(b * 16 + s)) * 64 * 1024;

    __shared__ __align__(16) float qs[1024];
    __shared__ float attn_s[64 * 16];      // [valid][head]
    const int t = threadIdx.x;

    for (int i = t; i < 1024; i += 256) qs[i] = qr[i];
    __syncthreads();

    const float scale = 0.125f;            // 1/sqrt(d_k=64)
    for (int p = t; p < 1024; p += 256) {
        const int vi = p >> 4;             // valid 0..63
        const int h  = p & 15;             // head  0..15
        const float4* k4 = reinterpret_cast<const float4*>(kr + (size_t)vi * 1024 + h * 64);
        const float4* q4 = reinterpret_cast<const float4*>(qs + h * 64);
        float dot = 0.0f;
        #pragma unroll
        for (int d = 0; d < 16; ++d) {
            float4 a = q4[d], c = k4[d];
            dot += a.x * c.x + a.y * c.y + a.z * c.z + a.w * c.w;
        }
        attn_s[p] = 1.0f / (1.0f + __expf(-dot * scale));
    }
    __syncthreads();

    if (t < 64) {
        float sum = 0.0f;
        #pragma unroll
        for (int h = 0; h < 16; ++h) sum += attn_s[t * 16 + h];
        attn_mean[(size_t)word * 64 + t] = sum * (1.0f / 16.0f);
    }

    for (int p = t; p < 1024; p += 256) {  // p = h*64 + d
        const int h = p >> 6;
        float acc = 0.0f;
        #pragma unroll 8
        for (int vi = 0; vi < 64; ++vi)
            acc += attn_s[vi * 16 + h] * vr[(size_t)vi * 1024 + p];
        out[(size_t)word * 1024 + p] = acc;
    }
}

// ---------------------------------------------------------------------------
extern "C" void kernel_launch(void* const* d_in, const int* in_sizes, int n_in,
                              void* d_out, int out_size, void* d_ws, size_t ws_size,
                              hipStream_t stream) {
    // setup_inputs order:
    // 0 text, 1 av_feat, 2 attn_sent_index, 3 Wq, 4 bq, 5 Wk, 6 bk,
    // 7 Wv, 8 bv, 9 g_text, 10 b_text, 11 g_av, 12 b_av
    const float* text    = (const float*)d_in[0];
    const float* av_feat = (const float*)d_in[1];
    const int*   sidx    = (const int*)  d_in[2];
    const float* Wq = (const float*)d_in[3];
    const float* bq = (const float*)d_in[4];
    const float* Wk = (const float*)d_in[5];
    const float* bk = (const float*)d_in[6];
    const float* Wv = (const float*)d_in[7];
    const float* bv = (const float*)d_in[8];
    const float* g_text = (const float*)d_in[9];
    const float* b_text = (const float*)d_in[10];
    const float* g_av   = (const float*)d_in[11];
    const float* b_av   = (const float*)d_in[12];

    const int M_TEXT = 4 * 512;        // 2048 rows
    const int M_AV   = 4 * 16 * 64;    // 4096 rows
    const int D      = 1024;

    // workspace layout (MiB-aligned chunks)
    char* ws = (char*)d_ws;
    unsigned short* text_bf = (unsigned short*)(ws);                        //  4 MiB
    unsigned short* av_bf   = (unsigned short*)(ws + (4u  << 20));          //  8 MiB
    unsigned short* wq_bf   = (unsigned short*)(ws + (12u << 20));          //  2 MiB
    unsigned short* wk_bf   = (unsigned short*)(ws + (14u << 20));          //  2 MiB
    unsigned short* wv_bf   = (unsigned short*)(ws + (16u << 20));          //  2 MiB
    float*          qbuf    = (float*)         (ws + (18u << 20));          //  8 MiB
    float*          kbuf    = (float*)         (ws + (26u << 20));          // 16 MiB
    float*          vbuf    = (float*)         (ws + (42u << 20));          // 16 MiB

    // 1) weight repack f32 -> bf16  (1M elems each = 262144 float4s)
    {
        dim3 g(1024), blk(256);
        cvt_f32_bf16_kernel<<<g, blk, 0, stream>>>(Wq, wq_bf, 262144);
        cvt_f32_bf16_kernel<<<g, blk, 0, stream>>>(Wk, wk_bf, 262144);
        cvt_f32_bf16_kernel<<<g, blk, 0, stream>>>(Wv, wv_bf, 262144);
    }

    // 2) layernorm -> bf16
    layernorm_bf16_kernel<<<dim3(M_TEXT), dim3(256), 0, stream>>>(text, g_text, b_text, text_bf);
    layernorm_bf16_kernel<<<dim3(M_AV),   dim3(256), 0, stream>>>(av_feat, g_av, b_av, av_bf);

    // 3) projections via bf16 WMMA (block = 128M x 64N, async-LDS staged A+B)
    gemm_bf16_wmma_kernel<<<dim3(M_TEXT / 128, 16), dim3(256), 0, stream>>>(text_bf, wq_bf, bq, qbuf);
    gemm_bf16_wmma_kernel<<<dim3(M_AV   / 128, 16), dim3(256), 0, stream>>>(av_bf,   wk_bf, bk, kbuf);
    gemm_bf16_wmma_kernel<<<dim3(M_AV   / 128, 16), dim3(256), 0, stream>>>(av_bf,   wv_bf, bv, vbuf);

    // 4) gathered sigmoid attention
    float* out_main = (float*)d_out;                       // 2048*1024
    float* out_mean = out_main + (size_t)M_TEXT * D;       // 2048*64
    attention_kernel<<<dim3(M_TEXT), dim3(256), 0, stream>>>(qbuf, kbuf, vbuf, sidx,
                                                             out_main, out_mean);
    (void)in_sizes; (void)n_in; (void)out_size; (void)ws_size;
}
